// Denoise_48507360641325
// MI455X (gfx1250) — compile-verified, matrix operand order checked by reading, not run on hardware
//
#include <hip/hip_runtime.h>

#define N_NODES 50000
#define N_EDGES 800000
#define LAYERS  4

typedef __attribute__((ext_vector_type(16))) __bf16 v16bf;
typedef __attribute__((ext_vector_type(8)))  __bf16 v8bf;
typedef __attribute__((ext_vector_type(2)))  __bf16 v2bf;
typedef __attribute__((ext_vector_type(8)))  float  v8f;

#if defined(__has_builtin) && __has_builtin(__builtin_amdgcn_cvt_pk_bf16_f32)
__device__ __forceinline__ v2bf cvt2(float a, float b) {
    return __builtin_amdgcn_cvt_pk_bf16_f32(a, b);
}
#else
__device__ __forceinline__ v2bf cvt2(float a, float b) {
    v2bf r; r[0] = (__bf16)a; r[1] = (__bf16)b; return r;
}
#endif
__device__ __forceinline__ __bf16 f2bf(float f) { return cvt2(f, f)[0]; }

// 16-bit A/B fragment lane map (ISA 7.12.2): lane L holds row/col (L&15),
// halves 0..7 -> K = kb+h, halves 8..15 -> K = 16+kb+(h-8), kb = 8*(L>=16).
__device__ __forceinline__ int frag_k(int lane, int h) {
    int kb = (lane >= 16) ? 8 : 0;
    return (h < 8) ? (kb + h) : (16 + kb + (h - 8));
}

// fast SiLU: v * rcp(1 + exp(-v)); v_rcp_f32 is ~1 ulp, fine for activations
__device__ __forceinline__ float silu_f(float v) {
    return v * __builtin_amdgcn_rcpf(1.0f + __expf(-v));
}

// A fragment from two 8-element bf16 runs (16 B each -> b128 loads, no cvt).
__device__ __forceinline__ v16bf load_a_bf(const __bf16* p0, const __bf16* p1) {
    union { v16bf v; v8bf h[2]; } u;
    u.h[0] = *(const v8bf*)p0;
    u.h[1] = *(const v8bf*)p1;
    return u.v;
}
// A fragment from two 8-float runs, scaled, packed-converted.
__device__ __forceinline__ v16bf load_a_f32s(const float* p0, const float* p1, float sc) {
    union { v16bf v; v2bf h2[8]; } u;
#pragma unroll
    for (int i = 0; i < 4; i++) u.h2[i]     = cvt2(p0[2*i] * sc, p0[2*i+1] * sc);
#pragma unroll
    for (int i = 0; i < 4; i++) u.h2[4 + i] = cvt2(p1[2*i] * sc, p1[2*i+1] * sc);
    return u.v;
}

// unpack adjacent bf16 pair (one dword) to two floats
__device__ __forceinline__ void bf2_unpack(unsigned u, float& f0, float& f1) {
    f0 = __builtin_bit_cast(float, u << 16);
    f1 = __builtin_bit_cast(float, u & 0xFFFF0000u);
}

#define WMMA_BF16(A, B, C) \
    __builtin_amdgcn_wmma_f32_16x16x32_bf16(false, (A), false, (B), (short)0, (C), false, false)

// Convert a KxN (N=64) fp32 weight matrix into bf16 B-fragments in LDS.
__device__ __forceinline__ void prep_frags(const float* __restrict__ W,
                                           __bf16 (*dst)[4][32][16],
                                           int nkt, int tid, int nthr) {
    int tot = nkt * 4 * 32 * 16;
    for (int idx = tid; idx < tot; idx += nthr) {
        int hh = idx & 15, ln = (idx >> 4) & 31, ct = (idx >> 9) & 3, kt = idx >> 11;
        int k = kt * 32 + frag_k(ln, hh);
        int n = ct * 16 + (ln & 15);
        dst[kt][ct][ln][hh] = f2bf(W[k * 64 + n]);
    }
}

// ---------------------------------------------------------------------------
// Fused per-edge pipeline: edge MLP -> message m, pos weight w,
// atomic scatters into x_acc, m_sum. One wave per 16-edge tile.
// ---------------------------------------------------------------------------
#define EDGE_WAVES 4
__global__ void __launch_bounds__(32 * EDGE_WAVES) egnn_edge(
    const float* __restrict__ x, const __bf16* __restrict__ hbf,
    const int* __restrict__ esrc, const int* __restrict__ edst,
    const float* __restrict__ eW1, const float* __restrict__ eb1,
    const float* __restrict__ eW2, const float* __restrict__ eb2,
    const float* __restrict__ pW1, const float* __restrict__ pb1,
    const float* __restrict__ pW2, const float* __restrict__ pb2v,
    float* __restrict__ xacc, float* __restrict__ msum,
    int numTiles)
{
    __shared__ __align__(32) __bf16 sW1[4][4][32][16];   // eW1 (K=128)
    __shared__ __align__(32) __bf16 sW2[2][4][32][16];   // eW2 (K=64)
    __shared__ __align__(32) __bf16 sP1[2][4][32][16];   // pW1 (K=64)
    __shared__ float sW1last[64], sEb1[64], sEb2[64], sPb1[64], sPw2[64];
    __shared__ float sPb2;
    __shared__ __align__(16) __bf16 stage[EDGE_WAVES][16 * 64];
    __shared__ float dlds[EDGE_WAVES][16];
    __shared__ float wlds[EDGE_WAVES][16];
    __shared__ int   slds[EDGE_WAVES][16];

    int tid = threadIdx.x, nthr = blockDim.x;
    prep_frags(eW1, sW1, 4, tid, nthr);
    prep_frags(eW2, sW2, 2, tid, nthr);
    prep_frags(pW1, sP1, 2, tid, nthr);
    for (int c = tid; c < 64; c += nthr) {
        sW1last[c] = eW1[128 * 64 + c];   // row 128 of eW1: the d_ij column
        sEb1[c] = eb1[c]; sEb2[c] = eb2[c]; sPb1[c] = pb1[c]; sPw2[c] = pW2[c];
    }
    if (tid == 0) sPb2 = pb2v[0];
    __syncthreads();

    int w = tid >> 5, lane = tid & 31;
    int r = lane & 15, hi = lane >> 4;
    int kb = hi * 8;
    __bf16* sp = &stage[w][0];

    for (int tile = blockIdx.x * EDGE_WAVES + w; tile < numTiles;
         tile += gridDim.x * EDGE_WAVES) {
        int e = tile * 16 + r;
        int s = esrc[e], dt = edst[e];
        float dx0 = x[dt * 3 + 0] - x[s * 3 + 0];
        float dx1 = x[dt * 3 + 1] - x[s * 3 + 1];
        float dx2 = x[dt * 3 + 2] - x[s * 3 + 2];
        float d = sqrtf(dx0 * dx0 + dx1 * dx1 + dx2 * dx2);
        if (hi == 0) { dlds[w][r] = d; slds[w][r] = s; }

        // A fragments of inp = [h_src | h_dst] (K=0..127): direct bf16 gathers
        const __bf16* hs = hbf + (size_t)s * 64;
        const __bf16* hd = hbf + (size_t)dt * 64;
        v16bf a0 = load_a_bf(hs + kb,      hs + 16 + kb);
        v16bf a1 = load_a_bf(hs + 32 + kb, hs + 48 + kb);
        v16bf a2 = load_a_bf(hd + kb,      hd + 16 + kb);
        v16bf a3 = load_a_bf(hd + 32 + kb, hd + 48 + kb);

        // broadcast d for the 8 rows this lane accumulates
        float dloc[8];
#pragma unroll
        for (int j = 0; j < 8; j++) dloc[j] = dlds[w][j + 8 * hi];

        // GEMM1 (K=128) + d_ij rank-1 column + bias + SiLU -> stage (bf16)
#pragma unroll
        for (int ct = 0; ct < 4; ct++) {
            v8f acc = {0, 0, 0, 0, 0, 0, 0, 0};
            acc = WMMA_BF16(a0, *(const v16bf*)&sW1[0][ct][lane][0], acc);
            acc = WMMA_BF16(a1, *(const v16bf*)&sW1[1][ct][lane][0], acc);
            acc = WMMA_BF16(a2, *(const v16bf*)&sW1[2][ct][lane][0], acc);
            acc = WMMA_BF16(a3, *(const v16bf*)&sW1[3][ct][lane][0], acc);
            int col = ct * 16 + r;
            float wl = sW1last[col], b1 = sEb1[col];
#pragma unroll
            for (int j = 0; j < 8; j++)
                sp[(j + 8 * hi) * 64 + col] = f2bf(silu_f(acc[j] + dloc[j] * wl + b1));
        }

        // transpose round-trip: A fragments (K=64) straight from bf16 stage
        v16bf c0 = load_a_bf(sp + r * 64 + kb,      sp + r * 64 + 16 + kb);
        v16bf c1 = load_a_bf(sp + r * 64 + 32 + kb, sp + r * 64 + 48 + kb);

        // GEMM2 + bias + SiLU -> messages m into stage
#pragma unroll
        for (int ct = 0; ct < 4; ct++) {
            v8f acc = {0, 0, 0, 0, 0, 0, 0, 0};
            acc = WMMA_BF16(c0, *(const v16bf*)&sW2[0][ct][lane][0], acc);
            acc = WMMA_BF16(c1, *(const v16bf*)&sW2[1][ct][lane][0], acc);
            int col = ct * 16 + r;
            float b2 = sEb2[col];
#pragma unroll
            for (int j = 0; j < 8; j++)
                sp[(j + 8 * hi) * 64 + col] = f2bf(silu_f(acc[j] + b2));
        }

        v16bf m0 = load_a_bf(sp + r * 64 + kb,      sp + r * 64 + 16 + kb);
        v16bf m1 = load_a_bf(sp + r * 64 + 32 + kb, sp + r * 64 + 48 + kb);

        // GEMM3 (pW1) + bias + SiLU, fused with 64->1 pW2 dot product
        float wacc[8] = {0, 0, 0, 0, 0, 0, 0, 0};
#pragma unroll
        for (int ct = 0; ct < 4; ct++) {
            v8f acc = {0, 0, 0, 0, 0, 0, 0, 0};
            acc = WMMA_BF16(m0, *(const v16bf*)&sP1[0][ct][lane][0], acc);
            acc = WMMA_BF16(m1, *(const v16bf*)&sP1[1][ct][lane][0], acc);
            int col = ct * 16 + r;
            float b3 = sPb1[col], pw = sPw2[col];
#pragma unroll
            for (int j = 0; j < 8; j++)
                wacc[j] += silu_f(acc[j] + b3) * pw;
        }
        // reduce over the 16 column-lanes of each half (rows j+8*hi)
#pragma unroll
        for (int msk = 1; msk < 16; msk <<= 1)
#pragma unroll
            for (int j = 0; j < 8; j++)
                wacc[j] += __shfl_xor(wacc[j], msk, 16);
        if (r == 0) {
#pragma unroll
            for (int j = 0; j < 8; j++) wlds[w][8 * hi + j] = wacc[j] + sPb2;
        }

        // scatter: position update (one edge per lane 0..15)
        if (hi == 0) {
            float wv = wlds[w][r];
            unsafeAtomicAdd(&xacc[s * 3 + 0], dx0 * wv);
            unsafeAtomicAdd(&xacc[s * 3 + 1], dx1 * wv);
            unsafeAtomicAdd(&xacc[s * 3 + 2], dx2 * wv);
        }
        // scatter: messages. Lane covers adjacent cols (2*lane, 2*lane+1):
        // one ds_load_b32 + bit-unpack; row index is wave-uniform -> SALU addr.
#pragma unroll
        for (int rr = 0; rr < 16; rr++) {
            int ss = __builtin_amdgcn_readfirstlane(slds[w][rr]);
            unsigned u = *(const unsigned*)(sp + rr * 64 + 2 * lane);
            float f0, f1;
            bf2_unpack(u, f0, f1);
            float* base = msum + (size_t)ss * 64 + 2 * lane;
            unsafeAtomicAdd(base,     f0);
            unsafeAtomicAdd(base + 1, f1);
        }
    }
}

// ---------------------------------------------------------------------------
// Fused per-node update: h += MLP([h, m_i]), x += x_acc. One wave per 16 nodes.
// Maintains the bf16 mirror of h.
// ---------------------------------------------------------------------------
#define NODE_WAVES 4
__global__ void __launch_bounds__(32 * NODE_WAVES) egnn_node(
    float* __restrict__ x, float* __restrict__ h, __bf16* __restrict__ hbf,
    const float* __restrict__ xacc, const float* __restrict__ msum,
    const float* __restrict__ cnt,
    const float* __restrict__ nW1, const float* __restrict__ nb1,
    const float* __restrict__ nW2, const float* __restrict__ nb2,
    int numTiles)
{
    __shared__ __align__(32) __bf16 sN1[4][4][32][16];   // nW1 (K=128)
    __shared__ __align__(32) __bf16 sN2[2][4][32][16];   // nW2 (K=64)
    __shared__ float sB1[64], sB2[64];
    __shared__ __align__(16) __bf16 stage[NODE_WAVES][16 * 64];

    int tid = threadIdx.x, nthr = blockDim.x;
    prep_frags(nW1, sN1, 4, tid, nthr);
    prep_frags(nW2, sN2, 2, tid, nthr);
    for (int c = tid; c < 64; c += nthr) { sB1[c] = nb1[c]; sB2[c] = nb2[c]; }
    __syncthreads();

    int w = tid >> 5, lane = tid & 31;
    int r = lane & 15, hi = lane >> 4;
    int kb = hi * 8;
    __bf16* sp = &stage[w][0];

    for (int tile = blockIdx.x * NODE_WAVES + w; tile < numTiles;
         tile += gridDim.x * NODE_WAVES) {
        int node = tile * 16 + r;
        const __bf16* hr = hbf + (size_t)node * 64;
        const float*  mr = msum + (size_t)node * 64;
        float inv = 1.0f / fmaxf(cnt[node], 1.0f);

        v16bf a0 = load_a_bf(hr + kb,      hr + 16 + kb);
        v16bf a1 = load_a_bf(hr + 32 + kb, hr + 48 + kb);
        v16bf a2 = load_a_f32s(mr + kb,      mr + 16 + kb, inv);
        v16bf a3 = load_a_f32s(mr + 32 + kb, mr + 48 + kb, inv);

        // GEMM1 (K=128) + bias + SiLU -> stage (bf16)
#pragma unroll
        for (int ct = 0; ct < 4; ct++) {
            v8f acc = {0, 0, 0, 0, 0, 0, 0, 0};
            acc = WMMA_BF16(a0, *(const v16bf*)&sN1[0][ct][lane][0], acc);
            acc = WMMA_BF16(a1, *(const v16bf*)&sN1[1][ct][lane][0], acc);
            acc = WMMA_BF16(a2, *(const v16bf*)&sN1[2][ct][lane][0], acc);
            acc = WMMA_BF16(a3, *(const v16bf*)&sN1[3][ct][lane][0], acc);
            int col = ct * 16 + r;
            float b1 = sB1[col];
#pragma unroll
            for (int j = 0; j < 8; j++)
                sp[(j + 8 * hi) * 64 + col] = f2bf(silu_f(acc[j] + b1));
        }

        v16bf c0 = load_a_bf(sp + r * 64 + kb,      sp + r * 64 + 16 + kb);
        v16bf c1 = load_a_bf(sp + r * 64 + 32 + kb, sp + r * 64 + 48 + kb);

        // GEMM2 (K=64) + bias -> dh; h += dh in fp32 master, refresh bf16 mirror
#pragma unroll
        for (int ct = 0; ct < 4; ct++) {
            v8f acc = {0, 0, 0, 0, 0, 0, 0, 0};
            acc = WMMA_BF16(c0, *(const v16bf*)&sN2[0][ct][lane][0], acc);
            acc = WMMA_BF16(c1, *(const v16bf*)&sN2[1][ct][lane][0], acc);
            int col = ct * 16 + r;
            float b2 = sB2[col];
#pragma unroll
            for (int j = 0; j < 8; j++) {
                size_t nm = (size_t)(tile * 16 + j + 8 * hi);
                float nv = h[nm * 64 + col] + acc[j] + b2;
                h[nm * 64 + col]   = nv;
                hbf[nm * 64 + col] = f2bf(nv);
            }
        }

        // x += x_acc
        if (hi == 0) {
#pragma unroll
            for (int k = 0; k < 3; k++)
                x[node * 3 + k] += xacc[node * 3 + k];
        }
    }
}

// ---------------------------------------------------------------------------
// init: h = emb[z] (fp32 + bf16 mirror), x = x_in
// ---------------------------------------------------------------------------
__global__ void egnn_init(const float* __restrict__ x_in, const int* __restrict__ z,
                          const float* __restrict__ emb,
                          float* __restrict__ xb, float* __restrict__ hb,
                          __bf16* __restrict__ hbf, int n)
{
    int idx = blockIdx.x * blockDim.x + threadIdx.x;
    if (idx < n * 64) {
        float v = emb[(size_t)z[idx >> 6] * 64 + (idx & 63)];
        hb[idx]  = v;
        hbf[idx] = f2bf(v);
    }
    if (idx < n * 3) xb[idx] = x_in[idx];
}

// in-degree of src nodes: layer-invariant, computed once
__global__ void egnn_cnt(const int* __restrict__ esrc, float* __restrict__ cnt, int e)
{
    int idx = blockIdx.x * blockDim.x + threadIdx.x;
    if (idx < e) unsafeAtomicAdd(&cnt[esrc[idx]], 1.0f);
}

extern "C" void kernel_launch(void* const* d_in, const int* in_sizes, int n_in,
                              void* d_out, int out_size, void* d_ws, size_t ws_size,
                              hipStream_t stream)
{
    const float* x_in = (const float*)d_in[0];
    const int*   z    = (const int*)d_in[1];
    const int*   edges= (const int*)d_in[2];
    const float* emb  = (const float*)d_in[3];
    const float* eW1  = (const float*)d_in[4];
    const float* eb1  = (const float*)d_in[5];
    const float* eW2  = (const float*)d_in[6];
    const float* eb2  = (const float*)d_in[7];
    const float* pW1  = (const float*)d_in[8];
    const float* pb1  = (const float*)d_in[9];
    const float* pW2  = (const float*)d_in[10];
    const float* pb2  = (const float*)d_in[11];
    const float* nW1  = (const float*)d_in[12];
    const float* nb1  = (const float*)d_in[13];
    const float* nW2  = (const float*)d_in[14];
    const float* nb2  = (const float*)d_in[15];

    // workspace (floats): x[N*3] | h[N*64] | xacc[N*3] | msum[N*64] | cnt[N] | hbf(bf16 N*64)
    float* ws   = (float*)d_ws;
    float* xb   = ws;
    float* hb   = xb + (size_t)N_NODES * 3;
    float* xacc = hb + (size_t)N_NODES * 64;
    float* msum = xacc + (size_t)N_NODES * 3;
    float* cntb = msum + (size_t)N_NODES * 64;
    __bf16* hbf = (__bf16*)(cntb + N_NODES);

    egnn_init<<<(N_NODES * 64 + 255) / 256, 256, 0, stream>>>(x_in, z, emb, xb, hb, hbf, N_NODES);
    hipMemsetAsync(cntb, 0, (size_t)N_NODES * sizeof(float), stream);
    egnn_cnt<<<(N_EDGES + 255) / 256, 256, 0, stream>>>(edges, cntb, N_EDGES);

    const int eTiles = N_EDGES / 16;   // 50000, exact
    const int nTiles = N_NODES / 16;   // 3125, exact
    for (int l = 0; l < LAYERS; l++) {
        // xacc and msum are contiguous -> one memset; cnt stays valid all layers
        hipMemsetAsync(xacc, 0,
                       (size_t)(N_NODES * 3 + N_NODES * 64) * sizeof(float), stream);
        egnn_edge<<<512, 32 * EDGE_WAVES, 0, stream>>>(
            xb, hbf, edges, edges + N_EDGES,
            eW1 + (size_t)l * 129 * 64, eb1 + (size_t)l * 64,
            eW2 + (size_t)l * 64 * 64,  eb2 + (size_t)l * 64,
            pW1 + (size_t)l * 64 * 64,  pb1 + (size_t)l * 64,
            pW2 + (size_t)l * 64,       pb2 + l,
            xacc, msum, eTiles);
        egnn_node<<<256, 32 * NODE_WAVES, 0, stream>>>(
            xb, hb, hbf, xacc, msum, cntb,
            nW1 + (size_t)l * 128 * 64, nb1 + (size_t)l * 64,
            nW2 + (size_t)l * 64 * 64,  nb2 + (size_t)l * 64, nTiles);
    }

    hipMemcpyAsync(d_out, xb, (size_t)N_NODES * 3 * sizeof(float),
                   hipMemcpyDeviceToDevice, stream);
}
